// SpatialGraphTransformer_89404039234013
// MI455X (gfx1250) — matrix operand order repaired and checked by reading, not compiled, not run on hardware
//
#include <hip/hip_runtime.h>
#include <hip/hip_bf16.h>
#include <math.h>

// ---------------- problem constants (from reference) ----------------
#define N_NODES 20000
#define G_GENES 2000
#define DMODEL  256
#define NLAYERS 4
#define NHEADS  8
#define NEDGES  160000
#define SEDIM   64
#define DHEAD   32
#define DFF     1024
#define LNEPS   1e-5f

// ---------------- WMMA types ----------------
typedef __attribute__((ext_vector_type(4)))  __bf16 bf16x4;
typedef __attribute__((ext_vector_type(8)))  __bf16 bf16x8;
typedef __attribute__((ext_vector_type(16))) __bf16 bf16x16;
typedef __attribute__((ext_vector_type(8)))  float  f32x8;

union FragAB { bf16x16 v; bf16x8 h[2]; };

// ---------------- GEMM tiling ----------------
#define BM 128
#define BN 128
#define BK 32
#define LDT 40   // padded LDS row (bf16 elems); 80B row stride -> conflict-free, 16B aligned

__device__ __forceinline__ float gelu_fn(float x) {
  const float c = 0.7978845608028654f;      // sqrt(2/pi)
  float t = c * (x + 0.044715f * x * x * x);
  return 0.5f * x * (1.0f + tanhf(t));
}

// C[M,Nn] = act(A[M,K] @ B[K,Nn] + bias);  A,B fp32 row-major, converted to bf16 in LDS.
// REQUIRES: K % 4 == 0 and Nn % 4 == 0 (true for all GEMMs in this network).
__global__ __launch_bounds__(256) void gemm_bf16_wmma(
    const float* __restrict__ A, const float* __restrict__ B,
    const float* __restrict__ bias, float* __restrict__ C,
    int M, int Nn, int K, int act)
{
  __shared__ __bf16 As[BM * LDT];   // As[row][k]
  __shared__ __bf16 Bs[BN * LDT];   // Bs[col][k]  (B transposed in LDS)

  const int tid  = threadIdx.x;
  const int lane = tid & 31;
  const int wid  = tid >> 5;
  const int wr   = wid >> 2;        // 0..1  (wave row)
  const int wc   = wid & 3;         // 0..3  (wave col)
  const int rowBlock = blockIdx.y * BM;
  const int colBlock = blockIdx.x * BN;

  f32x8 acc[4][2];
#pragma unroll
  for (int i = 0; i < 4; ++i)
#pragma unroll
    for (int j = 0; j < 2; ++j)
#pragma unroll
      for (int e = 0; e < 8; ++e) acc[i][j][e] = 0.0f;

  // staging thread mapping (loop-invariant)
  const int a_r0 = tid >> 3;            // 0..31
  const int a_c4 = (tid & 7) << 2;      // 0..28
  const int b_n4 = (tid & 31) << 2;     // 0..124
  const int b_kr = tid >> 5;            // 0..7
  const int b_gn = colBlock + b_n4;

  const int kTiles = (K + BK - 1) / BK;
  for (int kt = 0; kt < kTiles; ++kt) {
    const int k0 = kt * BK;

    // ---- stage A tile (BM x BK): branch-free clamped loads, then convert ----
    float4 fa[4];
    bool   oka[4];
    {
      const int gk = k0 + a_c4;
#pragma unroll
      for (int it = 0; it < 4; ++it) {
        const int gm = rowBlock + a_r0 + (it << 5);
        oka[it] = (gm < M) && (gk < K);                 // K%4==0 -> whole vec in range
        const int cm = oka[it] ? gm : 0;
        const int ck = oka[it] ? gk : 0;
        fa[it] = *(const float4*)(A + (size_t)cm * K + ck);   // always issued
      }
    }
    // ---- stage B tile (BK x BN): branch-free clamped loads ----
    float4 fb[4];
    bool   okb[4];
    {
#pragma unroll
      for (int it = 0; it < 4; ++it) {
        const int kk = (it << 3) + b_kr;                // 0..31
        const int gk = k0 + kk;
        okb[it] = (gk < K) && (b_gn < Nn);              // Nn%4==0 -> whole vec in range
        const int ck = okb[it] ? gk : 0;
        const int cn = okb[it] ? b_gn : 0;
        fb[it] = *(const float4*)(B + (size_t)ck * Nn + cn);  // always issued
      }
    }
    // convert + LDS stores (loads above can overlap; single wait here)
#pragma unroll
    for (int it = 0; it < 4; ++it) {
      float4 f = fa[it];
      if (!oka[it]) f = make_float4(0.f, 0.f, 0.f, 0.f);
      bf16x4 pk = { (__bf16)f.x, (__bf16)f.y, (__bf16)f.z, (__bf16)f.w };
      *(bf16x4*)(&As[(a_r0 + (it << 5)) * LDT + a_c4]) = pk;  // 8B aligned
    }
#pragma unroll
    for (int it = 0; it < 4; ++it) {
      float4 f = fb[it];
      if (!okb[it]) f = make_float4(0.f, 0.f, 0.f, 0.f);
      const int kk = (it << 3) + b_kr;
      Bs[(b_n4 + 0) * LDT + kk] = (__bf16)f.x;
      Bs[(b_n4 + 1) * LDT + kk] = (__bf16)f.y;
      Bs[(b_n4 + 2) * LDT + kk] = (__bf16)f.z;
      Bs[(b_n4 + 3) * LDT + kk] = (__bf16)f.w;
    }
    __syncthreads();

    // ---- fragments: ISA 7.12.2 layouts ----
    // A 16x32: m = lane&15; k = (i/8)*16 + (lane>>4)*8 + (i%8)
    FragAB af[4];
#pragma unroll
    for (int i = 0; i < 4; ++i) {
      const int r = wr * 64 + i * 16 + (lane & 15);
      const __bf16* p = &As[r * LDT + ((lane >> 4) << 3)];
      af[i].h[0] = *(const bf16x8*)(p);        // K = kb .. kb+7
      af[i].h[1] = *(const bf16x8*)(p + 16);   // K = 16+kb .. 16+kb+7
    }
    // B 32x16: n = lane&15; k = (i/8)*16 + (lane>>4)*8 + (i%8)
    FragAB bfr[2];
#pragma unroll
    for (int j = 0; j < 2; ++j) {
      const int n = wc * 32 + j * 16 + (lane & 15);
      const __bf16* p = &Bs[n * LDT + ((lane >> 4) << 3)];
      bfr[j].h[0] = *(const bf16x8*)(p);
      bfr[j].h[1] = *(const bf16x8*)(p + 16);
    }
#pragma unroll
    for (int i = 0; i < 4; ++i)
#pragma unroll
      for (int j = 0; j < 2; ++j)
        acc[i][j] = __builtin_amdgcn_wmma_f32_16x16x32_bf16(
            false, af[i].v, false, bfr[j].v, (short)0, acc[i][j], false, false);
    __syncthreads();
  }

  // ---- epilogue: C/D layout m = (lane>>4)*8 + e, n = lane&15 ----
#pragma unroll
  for (int i = 0; i < 4; ++i) {
    const int gm0 = rowBlock + wr * 64 + i * 16 + ((lane >> 4) << 3);
#pragma unroll
    for (int j = 0; j < 2; ++j) {
      const int gn = colBlock + wc * 32 + j * 16 + (lane & 15);
      if (gn >= Nn) continue;
      const float bv = bias ? bias[gn] : 0.0f;
#pragma unroll
      for (int e = 0; e < 8; ++e) {
        float v = acc[i][j][e] + bv;
        if (act == 1) v = gelu_fn(v);         // single GELU expansion
        if (gm0 + e < M)                      // predicate only the store
          C[(size_t)(gm0 + e) * Nn + gn] = v;
      }
    }
  }
}

// ---------------- LayerNorm: out = LN(y (+res)) * g + b ; one block per row ----------------
__global__ __launch_bounds__(256) void layernorm_k(
    const float* __restrict__ y, const float* __restrict__ res,
    const float* __restrict__ g, const float* __restrict__ b,
    float* __restrict__ out)
{
  __shared__ float r1[256];
  __shared__ float r2[256];
  const int row = blockIdx.x;
  const int tid = threadIdx.x;
  const size_t idx = (size_t)row * DMODEL + tid;
  float v = y[idx];
  if (res) v += res[idx];
  r1[tid] = v; r2[tid] = v * v;
  __syncthreads();
  for (int s = 128; s > 0; s >>= 1) {
    if (tid < s) { r1[tid] += r1[tid + s]; r2[tid] += r2[tid + s]; }
    __syncthreads();
  }
  const float mean = r1[0] * (1.0f / DMODEL);
  const float var  = r2[0] * (1.0f / DMODEL) - mean * mean;
  const float rstd = rsqrtf(var + LNEPS);
  out[idx] = (v - mean) * rstd * g[tid] + b[tid];
}

// ---------------- Spatial PE + projection, added in place to x ----------------
__global__ __launch_bounds__(256) void spatial_k(
    const float* __restrict__ coords, const float* __restrict__ W_se,
    const float* __restrict__ b_se,  const float* __restrict__ W_sp,
    const float* __restrict__ b_sp,  float* __restrict__ x)
{
  __shared__ float sp[SEDIM];
  const int n   = blockIdx.x;
  const int tid = threadIdx.x;
  const float cx = coords[n * 2 + 0];
  const float cy = coords[n * 2 + 1];
  if (tid < 32) {
    sp[tid] = cx * W_se[tid] + cy * W_se[32 + tid] + b_se[tid];  // learned half
  } else if (tid < 64) {
    const int t2 = tid - 32;
    const int j  = t2 >> 1;
    const float dt  = expf((float)(2 * j) * (-9.210340371976184f / 32.0f)); // -ln(1e4)/half
    const float ang = cx * dt;
    sp[tid] = (t2 & 1) ? cosf(ang) : sinf(ang);
  }
  __syncthreads();
  float acc = b_sp[tid];
#pragma unroll 8
  for (int s = 0; s < SEDIM; ++s) acc += sp[s] * W_sp[s * DMODEL + tid];
  x[(size_t)n * DMODEL + tid] += acc;
}

// ---------------- scatter softmax helpers ----------------
__device__ __forceinline__ unsigned fenc(float f) {
  unsigned u = __float_as_uint(f);
  return (u & 0x80000000u) ? ~u : (u | 0x80000000u);
}
__device__ __forceinline__ float fdec(unsigned u) {
  u = (u & 0x80000000u) ? (u & 0x7FFFFFFFu) : ~u;
  return __uint_as_float(u);
}

__global__ void fill_u32_k(unsigned* __restrict__ p, unsigned v, int n) {
  const int i = blockIdx.x * blockDim.x + threadIdx.x;
  if (i < n) p[i] = v;
}
__global__ void fill_f32_k(float* __restrict__ p, float v, int n) {
  const int i = blockIdx.x * blockDim.x + threadIdx.x;
  if (i < n) p[i] = v;
}

__global__ void edge_scores_k(const float* __restrict__ q, const float* __restrict__ k,
    const float* __restrict__ edge_attr, const long long* __restrict__ eidx,
    const float* __restrict__ W_eb, const float* __restrict__ b_eb,
    float* __restrict__ scores, unsigned* __restrict__ mred)
{
  const int idx = blockIdx.x * blockDim.x + threadIdx.x;
  if (idx >= NEDGES * NHEADS) return;
  const int e = idx / NHEADS;
  const int h = idx % NHEADS;
  const int src = (int)eidx[e];
  const int dst = (int)eidx[NEDGES + e];
  const float* qp = q + (size_t)dst * DMODEL + h * DHEAD;
  const float* kp = k + (size_t)src * DMODEL + h * DHEAD;
  float s = 0.0f;
#pragma unroll
  for (int i = 0; i < DHEAD; i += 4) {
    const float4 a  = *(const float4*)(qp + i);
    const float4 bb = *(const float4*)(kp + i);
    s += a.x * bb.x + a.y * bb.y + a.z * bb.z + a.w * bb.w;
  }
  s *= 0.17677669529663687f;                       // 1/sqrt(DH)
  s += edge_attr[e] * W_eb[h] + b_eb[h];
  scores[idx] = s;
  atomicMax(&mred[dst * NHEADS + h], fenc(s));
}

__global__ void finalize_m_k(unsigned* __restrict__ mred, float* __restrict__ sred) {
  const int idx = blockIdx.x * blockDim.x + threadIdx.x;
  if (idx >= N_NODES * NHEADS) return;
  float m = fdec(mred[idx]);
  if (!(m >= -3.0e38f && m <= 3.0e38f)) m = 0.0f;  // non-finite -> 0 (matches reference)
  ((float*)mred)[idx] = m;
  sred[idx] = 0.0f;
}

__global__ void edge_exp_k(float* __restrict__ scores, const float* __restrict__ m,
    const long long* __restrict__ eidx, float* __restrict__ sred)
{
  const int idx = blockIdx.x * blockDim.x + threadIdx.x;
  if (idx >= NEDGES * NHEADS) return;
  const int e = idx / NHEADS;
  const int h = idx % NHEADS;
  const int dst = (int)eidx[NEDGES + e];
  const float ev = expf(scores[idx] - m[dst * NHEADS + h]);
  scores[idx] = ev;
  atomicAdd(&sred[dst * NHEADS + h], ev);
}

__global__ void edge_agg_k(const float* __restrict__ scores, const float* __restrict__ sred,
    const float* __restrict__ v, const long long* __restrict__ eidx, float* __restrict__ agg)
{
  const int idx = blockIdx.x * blockDim.x + threadIdx.x;
  if (idx >= NEDGES * NHEADS) return;
  const int e = idx / NHEADS;
  const int h = idx % NHEADS;
  const int src = (int)eidx[e];
  const int dst = (int)eidx[NEDGES + e];
  const float alpha = scores[idx] / (sred[dst * NHEADS + h] + 1e-16f);
  const float* vp = v + (size_t)src * DMODEL + h * DHEAD;
  float* ap = agg + (size_t)dst * DMODEL + h * DHEAD;
#pragma unroll
  for (int i = 0; i < DHEAD; i += 4) {
    const float4 vv = *(const float4*)(vp + i);
    atomicAdd(&ap[i + 0], alpha * vv.x);
    atomicAdd(&ap[i + 1], alpha * vv.y);
    atomicAdd(&ap[i + 2], alpha * vv.z);
    atomicAdd(&ap[i + 3], alpha * vv.w);
  }
}

// ---------------- host orchestration ----------------
extern "C" void kernel_launch(void* const* d_in, const int* in_sizes, int n_in,
                              void* d_out, int out_size, void* d_ws, size_t ws_size,
                              hipStream_t stream)
{
  (void)in_sizes; (void)n_in; (void)out_size; (void)ws_size;
  const float* gene    = (const float*)d_in[0];
  const float* coords  = (const float*)d_in[1];
  const float* eattr   = (const float*)d_in[2];
  const float* W_ge    = (const float*)d_in[3];
  const float* b_ge    = (const float*)d_in[4];
  const float* ln_ge_g = (const float*)d_in[5];
  const float* ln_ge_b = (const float*)d_in[6];
  const float* W_se    = (const float*)d_in[7];
  const float* b_se    = (const float*)d_in[8];
  const float* W_sp    = (const float*)d_in[9];
  const float* b_sp    = (const float*)d_in[10];
  const float* Wq      = (const float*)d_in[11];
  const float* bq      = (const float*)d_in[12];
  const float* Wk      = (const float*)d_in[13];
  const float* bk      = (const float*)d_in[14];
  const float* Wv      = (const float*)d_in[15];
  const float* bv      = (const float*)d_in[16];
  const float* Wo      = (const float*)d_in[17];
  const float* bo      = (const float*)d_in[18];
  const float* W_eb    = (const float*)d_in[19];
  const float* b_eb    = (const float*)d_in[20];
  const float* W1      = (const float*)d_in[21];
  const float* b1      = (const float*)d_in[22];
  const float* W2      = (const float*)d_in[23];
  const float* b2      = (const float*)d_in[24];
  const float* ln1_g   = (const float*)d_in[25];
  const float* ln1_b   = (const float*)d_in[26];
  const float* ln2_g   = (const float*)d_in[27];
  const float* ln2_b   = (const float*)d_in[28];
  const float* fn_g    = (const float*)d_in[29];
  const float* fn_b    = (const float*)d_in[30];
  const long long* eidx = (const long long*)d_in[31];
  float* out = (float*)d_out;

  // workspace layout (fp32 elements)
  float* ws = (float*)d_ws;
  const size_t ND = (size_t)N_NODES * DMODEL;
  float* xA   = ws;
  float* xB   = xA  + ND;
  float* qb   = xB  + ND;
  float* kb   = qb  + ND;
  float* vb   = kb  + ND;
  float* tmp  = vb  + ND;
  float* agg  = tmp + ND;
  float* h1   = agg + ND;                       // N * DFF
  float* sc   = h1  + (size_t)N_NODES * DFF;    // E * H
  float* mred = sc  + (size_t)NEDGES * NHEADS;  // N * H (uint view during max)
  float* sred = mred + (size_t)N_NODES * NHEADS;

  const dim3 blk(256);
  auto gemm = [&](const float* A, const float* B, const float* bias, float* C,
                  int M, int Nn, int K, int act) {
    dim3 grid((Nn + BN - 1) / BN, (M + BM - 1) / BM);
    gemm_bf16_wmma<<<grid, blk, 0, stream>>>(A, B, bias, C, M, Nn, K, act);
  };
  const int EH  = NEDGES * NHEADS;
  const int NH  = N_NODES * NHEADS;
  const int ehB = (EH + 255) / 256;
  const int nhB = (NH + 255) / 256;
  const int ndB = (int)((ND + 255) / 256);

  // ---- encoder: x = LN(gene @ W_ge + b_ge); x += spatialPE(coords) @ W_sp + b_sp ----
  gemm(gene, W_ge, b_ge, tmp, N_NODES, DMODEL, G_GENES, 0);
  layernorm_k<<<N_NODES, blk, 0, stream>>>(tmp, nullptr, ln_ge_g, ln_ge_b, xA);
  spatial_k<<<N_NODES, blk, 0, stream>>>(coords, W_se, b_se, W_sp, b_sp, xA);

  float* x  = xA;
  float* xo = xB;
  for (int l = 0; l < NLAYERS; ++l) {
    const size_t DD = (size_t)DMODEL * DMODEL;
    // Q,K,V projections
    gemm(x, Wq + l * DD, bq + l * DMODEL, qb, N_NODES, DMODEL, DMODEL, 0);
    gemm(x, Wk + l * DD, bk + l * DMODEL, kb, N_NODES, DMODEL, DMODEL, 0);
    gemm(x, Wv + l * DD, bv + l * DMODEL, vb, N_NODES, DMODEL, DMODEL, 0);
    // scatter softmax over edges
    fill_u32_k<<<nhB, blk, 0, stream>>>((unsigned*)mred, 0x007FFFFFu /* enc(-inf) */, NH);
    fill_f32_k<<<ndB, blk, 0, stream>>>(agg, 0.0f, (int)ND);
    edge_scores_k<<<ehB, blk, 0, stream>>>(qb, kb, eattr, eidx,
                                           W_eb + l * NHEADS, b_eb + l * NHEADS,
                                           sc, (unsigned*)mred);
    finalize_m_k<<<nhB, blk, 0, stream>>>((unsigned*)mred, sred);
    edge_exp_k<<<ehB, blk, 0, stream>>>(sc, mred, eidx, sred);
    edge_agg_k<<<ehB, blk, 0, stream>>>(sc, sred, vb, eidx, agg);
    // output projection + residual LN
    gemm(agg, Wo + l * DD, bo + l * DMODEL, tmp, N_NODES, DMODEL, DMODEL, 0);
    layernorm_k<<<N_NODES, blk, 0, stream>>>(tmp, x, ln1_g + l * DMODEL, ln1_b + l * DMODEL, xo);
    { float* t = x; x = xo; xo = t; }
    // FFN
    gemm(x, W1 + (size_t)l * DMODEL * DFF, b1 + l * DFF, h1, N_NODES, DFF, DMODEL, 1);
    gemm(h1, W2 + (size_t)l * DFF * DMODEL, b2 + l * DMODEL, tmp, N_NODES, DMODEL, DFF, 0);
    layernorm_k<<<N_NODES, blk, 0, stream>>>(tmp, x, ln2_g + l * DMODEL, ln2_b + l * DMODEL, xo);
    { float* t = x; x = xo; xo = t; }
  }
  // final LN -> out
  layernorm_k<<<N_NODES, blk, 0, stream>>>(x, nullptr, fn_g, fn_b, out);
}
